// Qwen3MoeAttention_16690242912516
// MI455X (gfx1250) — matrix operand order repaired and checked by reading, not compiled, not run on hardware
//
#include <hip/hip_runtime.h>
#include <math.h>

// ---------------- problem constants ----------------
#define S_    2048
#define H_    2048
#define NH_   32
#define NKV_  4
#define D_    128
#define QKVN_ ((NH_ + 2 * NKV_) * D_)   // 5120
#define KOFF_ (NH_ * D_)                // 4096
#define VOFF_ ((NH_ + NKV_) * D_)       // 4608
#define GRP_  (NH_ / NKV_)              // 8

typedef __attribute__((ext_vector_type(16))) __bf16 v16bf;
typedef __attribute__((ext_vector_type(8)))  __bf16 v8bf;
typedef __attribute__((ext_vector_type(8)))  float  v8f;
typedef __attribute__((ext_vector_type(4)))  unsigned u32x4;
typedef __attribute__((ext_vector_type(8)))  unsigned u32x8;

__device__ __forceinline__ v8f vzero8() {
  v8f z = {0.f, 0.f, 0.f, 0.f, 0.f, 0.f, 0.f, 0.f};
  return z;
}

// ---- CDNA5 async global->LDS copy (ASYNCcnt) --------------------------
__device__ __forceinline__ void async_cp16(const void* g, void* l) {
  unsigned lds = (unsigned)(size_t)l;
  unsigned long long ga = (unsigned long long)(size_t)g;
  asm volatile("global_load_async_to_lds_b128 %0, %1, off"
               :: "v"(lds), "v"(ga) : "memory");
}
__device__ __forceinline__ void wait_async() {
  asm volatile("s_wait_asynccnt 0" ::: "memory");
}

// ---- CDNA5 Tensor Data Mover: 2-D tile global->LDS (TENSORcnt) --------
// D# per CDNA5 ISA (2-group form, bf16 elements).
// tile_d0 = elements per row, tile_d1 = rows, stride = row stride (elems).
__device__ __forceinline__ void tdm_load_2d(const void* gaddr, void* laddr,
                                            unsigned tile_d0, unsigned tile_d1,
                                            unsigned long long stride_elems) {
  unsigned long long ga = (unsigned long long)(size_t)gaddr;
  u32x4 g0;
  g0[0] = 1u;                                   // count=1, user descriptor
  g0[1] = (unsigned)(size_t)laddr;              // lds_addr (bytes)
  g0[2] = (unsigned)(ga & 0xffffffffu);         // global_addr[31:0]
  g0[3] = (unsigned)((ga >> 32) & 0x1ffffffu)   // global_addr[56:32]
          | (2u << 30);                         // type = 2 ("image")
  u32x8 g1;
  g1[0] = (1u << 16);                           // data_size=1 (2 bytes)
  g1[1] = (tile_d0 & 0xffffu) << 16;            // tensor_dim0[15:0]
  g1[2] = ((tile_d0 >> 16) & 0xffffu)           // tensor_dim0[31:16]
          | ((tile_d1 & 0xffffu) << 16);        // tensor_dim1[15:0]
  g1[3] = ((tile_d1 >> 16) & 0xffffu)           // tensor_dim1[31:16]
          | ((tile_d0 & 0xffffu) << 16);        // tile_dim0
  g1[4] = tile_d1 & 0xffffu;                    // tile_dim1 (tile_dim2=0)
  g1[5] = (unsigned)(stride_elems & 0xffffffffu);       // dim0_stride[31:0]
  g1[6] = (unsigned)((stride_elems >> 32) & 0xffffu);   // dim0_stride[47:32]
  g1[7] = 0u;                                   // dim1_stride unused (2-D)
  asm volatile("tensor_load_to_lds %0, %1" :: "s"(g0), "s"(g1) : "memory");
}
__device__ __forceinline__ void wait_tensor() {
  __builtin_amdgcn_s_wait_tensorcnt(0);
}

// Build a 16-element bf16 fragment from two 8-element halves at p and p+sec.
// sec==8 : 16 contiguous values; sec==16 : A- and B-fragment halves.
__device__ __forceinline__ v16bf ld_frag(const __bf16* p, int sec) {
  v8bf lo = *(const v8bf*)(p);
  v8bf hi = *(const v8bf*)(p + sec);
  v16bf r;
#pragma unroll
  for (int i = 0; i < 8; ++i) { r[i] = lo[i]; r[i + 8] = hi[i]; }
  return r;
}

// =====================================================================
// Bulk fp32 -> bf16 conversion, 4 elems/thread.
// =====================================================================
__global__ __launch_bounds__(256)
void f32_to_bf16(const float* __restrict__ x, __bf16* __restrict__ y, long n) {
  long i = ((long)blockIdx.x * blockDim.x + threadIdx.x) * 4;
  if (i + 3 < n) {
    float4 v = *(const float4*)(x + i);
    __bf16* d = y + i;
    d[0] = (__bf16)v.x; d[1] = (__bf16)v.y; d[2] = (__bf16)v.z; d[3] = (__bf16)v.w;
  }
}

// =====================================================================
// GEMM: Y[M,N] (f32) = X[M,K] (bf16) * W[K,N] (bf16)
// block = 256 (8 waves), tile 128x128x32, double-buffered LDS.
// A tiles: one TDM tensor_load_to_lds per stage (wave 0, TENSORcnt).
// B tiles: transposed to [n][k] via regular loads + ds stores.
// =====================================================================
__global__ __launch_bounds__(256)
void gemm_bf16(const __bf16* __restrict__ X, const __bf16* __restrict__ W,
               float* __restrict__ Y, int M, int N, int K) {
  __shared__ __bf16 As[2][128 * 32];   // [m][k], k contiguous
  __shared__ __bf16 Bs[2][128 * 32];   // [n][k], k contiguous

  const int tid  = threadIdx.x;
  const int lane = tid & 31;
  const int wv   = tid >> 5;
  const int col0 = lane & 15;
  const int rowOff = (lane < 16) ? 0 : 8;
  const int kbA = (lane < 16) ? 0 : 8;
  const int kbB = (lane < 16) ? 0 : 8;

  const int colBase = blockIdx.x * 128;
  const int rowBase = blockIdx.y * 128;

  v8f acc[8];
#pragma unroll
  for (int i = 0; i < 8; ++i) acc[i] = vzero8();

  auto stage = [&](int kb, int b) {
    // A tile 128x32: single TDM DMA issued by wave 0
    if (wv == 0)
      tdm_load_2d(X + (size_t)rowBase * K + kb, &As[b][0],
                  /*tile_d0=*/32, /*tile_d1=*/128,
                  /*stride=*/(unsigned long long)K);
    // B tile 32x128 -> transposed [n][k] (2 vec loads + 16 b16 stores)
#pragma unroll
    for (int i = 0; i < 2; ++i) {
      int idx = tid + i * 256;
      int k   = idx >> 4;
      int n8  = (idx & 15) << 3;
      v8bf v = *(const v8bf*)(W + (size_t)(kb + k) * N + colBase + n8);
#pragma unroll
      for (int j = 0; j < 8; ++j) Bs[b][(n8 + j) * 32 + k] = v[j];
    }
  };

  stage(0, 0);
  const int T = K >> 5;
  for (int t = 0; t < T; ++t) {
    const int b = t & 1;
    if (wv == 0) wait_tensor();
    __syncthreads();
    if (t + 1 < T) stage((t + 1) << 5, (t + 1) & 1);

    v16bf a = ld_frag(&As[b][(wv * 16 + col0) * 32 + kbA], 16);
#pragma unroll
    for (int nt = 0; nt < 8; ++nt) {
      v16bf bb = ld_frag(&Bs[b][(nt * 16 + col0) * 32 + kbB], 16);
      acc[nt] = __builtin_amdgcn_wmma_f32_16x16x32_bf16(
          false, a, false, bb, (short)0, acc[nt], false, false);
    }
  }

#pragma unroll
  for (int nt = 0; nt < 8; ++nt) {
#pragma unroll
    for (int r = 0; r < 8; ++r) {
      int row = rowBase + wv * 16 + r + rowOff;
      int col = colBase + nt * 16 + col0;
      Y[(size_t)row * N + col] = acc[nt][r];
    }
  }
}

// =====================================================================
// RMSNorm + RoPE (q, k) and plain convert (v), all emitted as bf16.
// grid = (S, NH + 2*NKV), block = 128.
// =====================================================================
__global__ __launch_bounds__(128)
void norm_rope(const int* __restrict__ positions, const float* __restrict__ qkv,
               const float* __restrict__ qw, const float* __restrict__ kw,
               __bf16* __restrict__ qr, __bf16* __restrict__ kr,
               __bf16* __restrict__ vb) {
  const int s = blockIdx.x;
  const int head = blockIdx.y;
  const int t = threadIdx.x;

  if (head >= NH_ + NKV_) {   // V: copy-convert only (uniform per block)
    int vh = head - NH_ - NKV_;
    float x = qkv[(size_t)s * QKVN_ + VOFF_ + vh * D_ + t];
    vb[((size_t)s * NKV_ + vh) * D_ + t] = (__bf16)x;
    return;
  }

  const bool isq = head < NH_;
  const float* w = isq ? qw : kw;
  size_t base = (size_t)s * QKVN_ + (isq ? head * D_ : KOFF_ + (head - NH_) * D_);
  float x = qkv[base + t];

  __shared__ float red[128];
  __shared__ float xs[128];
  red[t] = x * x;
  __syncthreads();
#pragma unroll
  for (int off = 64; off > 0; off >>= 1) {
    if (t < off) red[t] += red[t + off];
    __syncthreads();
  }
  float inv = rsqrtf(red[0] * (1.0f / D_) + 1e-6f);
  float xn = x * inv * w[t];
  xs[t] = xn;
  __syncthreads();

  const float pos = (float)positions[s];
  const float lt = logf(1000000.0f);
  float out;
  if (t < 64) {
    float fr = pos * __expf(-((float)t / 64.f) * lt);
    out = xs[t] * cosf(fr) - xs[t + 64] * sinf(fr);
  } else {
    int i = t - 64;
    float fr = pos * __expf(-((float)i / 64.f) * lt);
    out = xs[t] * cosf(fr) + xs[t - 64] * sinf(fr);
  }
  if (isq) qr[((size_t)s * NH_ + head) * D_ + t] = (__bf16)out;
  else     kr[((size_t)s * NKV_ + (head - NH_)) * D_ + t] = (__bf16)out;
}

// =====================================================================
// Flash attention (bf16 q/k/v, bf16 out): grid = (S/128, NH), block = 256.
// Double-buffered K/V chunks of 32 keys; K staged via async-to-LDS.
// =====================================================================
__global__ __launch_bounds__(256)
void attn_fa(const __bf16* __restrict__ qr, const __bf16* __restrict__ kr,
             const __bf16* __restrict__ vb, __bf16* __restrict__ attn) {
  __shared__ __bf16 Ks[2][32 * 128];    // [key][d]  (K^T as B)
  __shared__ __bf16 Vs[2][128 * 32];    // [d][key]  (V as B)
  __shared__ __bf16 Ps[8 * 16 * 32];    // per-wave P tile [row][key]

  const int tid  = threadIdx.x;
  const int lane = tid & 31;
  const int wv   = tid >> 5;
  const int col0 = lane & 15;
  const int rowOff = (lane < 16) ? 0 : 8;
  const int kbA = (lane < 16) ? 0 : 8;
  const int kbB = (lane < 16) ? 0 : 8;

  const int qtile = blockIdx.x;
  const int h     = blockIdx.y;
  const int kv    = h / GRP_;
  const int qbase = qtile * 128;
  const float scale = 0.088388347648318447f;   // 1/sqrt(128)

  // ---- preload Q fragments (4 d-chunks of 32) straight from bf16 global ----
  v16bf qf[4];
  {
    const int qrow = qbase + wv * 16 + col0;
    const __bf16* qp0 = qr + ((size_t)qrow * NH_ + h) * D_;
#pragma unroll
    for (int dc = 0; dc < 4; ++dc)
      qf[dc] = ld_frag(qp0 + dc * 32 + kbA, 16);
  }

  v8f o[8];
#pragma unroll
  for (int i = 0; i < 8; ++i) o[i] = vzero8();
  float m_run[8], l_run[8];
#pragma unroll
  for (int r = 0; r < 8; ++r) { m_run[r] = -1e30f; l_run[r] = 0.f; }

  auto stage = [&](int keyBase, int b) {
    // K chunk 32x128: async 16B copies (2 per thread, ASYNCcnt)
#pragma unroll
    for (int i = 0; i < 2; ++i) {
      int idx = tid + i * 256;
      int key = idx >> 4;
      int d8  = (idx & 15) << 3;
      async_cp16(kr + ((size_t)(keyBase + key) * NKV_ + kv) * D_ + d8,
                 &Ks[b][key * 128 + d8]);
    }
    // V chunk transposed -> [d][key]
#pragma unroll
    for (int i = 0; i < 2; ++i) {
      int idx = tid + i * 256;
      int key = idx >> 4;
      int d8  = (idx & 15) << 3;
      v8bf v = *(const v8bf*)(vb + ((size_t)(keyBase + key) * NKV_ + kv) * D_ + d8);
#pragma unroll
      for (int j = 0; j < 8; ++j) Vs[b][(d8 + j) * 32 + key] = v[j];
    }
  };

  const int nChunks = 4 * (qtile + 1);
  stage(0, 0);
  for (int c = 0; c < nChunks; ++c) {
    const int b = c & 1;
    const int keyBase = c * 32;
    wait_async();
    __syncthreads();
    if (c + 1 < nChunks) stage((c + 1) * 32, (c + 1) & 1);

    // ---- scores: S = Q * K^T (two 16-key tiles) ----
    v8f s0 = vzero8(), s1 = vzero8();
#pragma unroll
    for (int dc = 0; dc < 4; ++dc) {
      v16bf b0 = ld_frag(&Ks[b][(col0)      * 128 + dc * 32 + kbB], 16);
      v16bf b1 = ld_frag(&Ks[b][(16 + col0) * 128 + dc * 32 + kbB], 16);
      s0 = __builtin_amdgcn_wmma_f32_16x16x32_bf16(false, qf[dc], false, b0, (short)0, s0, false, false);
      s1 = __builtin_amdgcn_wmma_f32_16x16x32_bf16(false, qf[dc], false, b1, (short)0, s1, false, false);
    }

    // ---- online softmax (row across 16 lanes of a half-wave) ----
    float alphaA[8];
#pragma unroll
    for (int r = 0; r < 8; ++r) {
      const int qg = qbase + wv * 16 + r + rowOff;
      float x0 = s0[r] * scale;
      float x1 = s1[r] * scale;
      if (keyBase + col0      > qg) x0 = -1e30f;
      if (keyBase + 16 + col0 > qg) x1 = -1e30f;
      float mx = fmaxf(x0, x1);
      mx = fmaxf(mx, __shfl_xor(mx, 1));
      mx = fmaxf(mx, __shfl_xor(mx, 2));
      mx = fmaxf(mx, __shfl_xor(mx, 4));
      mx = fmaxf(mx, __shfl_xor(mx, 8));
      const float mnew = fmaxf(m_run[r], mx);
      const float p0 = __expf(x0 - mnew);
      const float p1 = __expf(x1 - mnew);
      alphaA[r] = __expf(m_run[r] - mnew);
      float rs = p0 + p1;
      rs += __shfl_xor(rs, 1);
      rs += __shfl_xor(rs, 2);
      rs += __shfl_xor(rs, 4);
      rs += __shfl_xor(rs, 8);
      l_run[r] = l_run[r] * alphaA[r] + rs;
      m_run[r] = mnew;
      const int prow = r + rowOff;
      Ps[wv * 512 + prow * 32 + col0]      = (__bf16)p0;
      Ps[wv * 512 + prow * 32 + 16 + col0] = (__bf16)p1;
    }
#pragma unroll
    for (int f = 0; f < 8; ++f)
#pragma unroll
      for (int r = 0; r < 8; ++r) o[f][r] *= alphaA[r];

    // ---- PV: O += P * V ----
    v16bf pf = ld_frag(&Ps[wv * 512 + col0 * 32 + kbA], 16);
#pragma unroll
    for (int f = 0; f < 8; ++f) {
      v16bf vf = ld_frag(&Vs[b][(f * 16 + col0) * 32 + kbB], 16);
      o[f] = __builtin_amdgcn_wmma_f32_16x16x32_bf16(false, pf, false, vf, (short)0, o[f], false, false);
    }
  }

  // ---- epilogue: bf16 attn[s][h*D + d] = o / l ----
#pragma unroll
  for (int f = 0; f < 8; ++f) {
#pragma unroll
    for (int r = 0; r < 8; ++r) {
      int row = qbase + wv * 16 + r + rowOff;
      attn[(size_t)row * (NH_ * D_) + h * D_ + f * 16 + col0] =
          (__bf16)(o[f][r] / l_run[r]);
    }
  }
}

// =====================================================================
extern "C" void kernel_launch(void* const* d_in, const int* in_sizes, int n_in,
                              void* d_out, int out_size, void* d_ws, size_t ws_size,
                              hipStream_t stream) {
  (void)in_sizes; (void)n_in; (void)out_size; (void)ws_size;
  const int*   positions = (const int*)d_in[0];
  const float* hidden    = (const float*)d_in[1];
  const float* w_qkv     = (const float*)d_in[2];
  const float* w_o       = (const float*)d_in[3];
  const float* qw        = (const float*)d_in[4];
  const float* kw        = (const float*)d_in[5];
  float* out = (float*)d_out;

  char* p = (char*)d_ws;
  float*  qkv   = (float*)p;   p += (size_t)S_ * QKVN_ * 4;        // fp32 qkv
  __bf16* hb    = (__bf16*)p;  p += (size_t)S_ * H_ * 2;           // hidden bf16
  __bf16* wqb   = (__bf16*)p;  p += (size_t)H_ * QKVN_ * 2;        // w_qkv bf16
  __bf16* wob   = (__bf16*)p;  p += (size_t)(NH_ * D_) * H_ * 2;   // w_o bf16
  __bf16* qrb   = (__bf16*)p;  p += (size_t)S_ * NH_ * D_ * 2;     // q (norm+rope)
  __bf16* krb   = (__bf16*)p;  p += (size_t)S_ * NKV_ * D_ * 2;    // k (norm+rope)
  __bf16* vbb   = (__bf16*)p;  p += (size_t)S_ * NKV_ * D_ * 2;    // v
  __bf16* attnb = (__bf16*)p;                                      // attn out bf16

  // 0) one-time operand conversions to bf16
  {
    long n1 = (long)S_ * H_;
    long n2 = (long)H_ * QKVN_;
    long n3 = (long)(NH_ * D_) * H_;
    f32_to_bf16<<<(unsigned)(n1 / 1024), 256, 0, stream>>>(hidden, hb, n1);
    f32_to_bf16<<<(unsigned)(n2 / 1024), 256, 0, stream>>>(w_qkv, wqb, n2);
    f32_to_bf16<<<(unsigned)(n3 / 1024), 256, 0, stream>>>(w_o, wob, n3);
  }
  // 1) QKV projection: [2048,2048] x [2048,5120] -> fp32 qkv
  gemm_bf16<<<dim3(QKVN_ / 128, S_ / 128), 256, 0, stream>>>(
      hb, wqb, qkv, S_, QKVN_, H_);
  // 2) RMSNorm + RoPE (q,k) and convert (v) -> bf16
  norm_rope<<<dim3(S_, NH_ + 2 * NKV_), 128, 0, stream>>>(
      positions, qkv, qw, kw, qrb, krb, vbb);
  // 3) causal flash attention (GQA) -> bf16 attn
  attn_fa<<<dim3(S_ / 128, NH_), 256, 0, stream>>>(qrb, krb, vbb, attnb);
  // 4) output projection: [2048,4096] x [4096,2048] -> fp32 out
  gemm_bf16<<<dim3(H_ / 128, S_ / 128), 256, 0, stream>>>(
      attnb, wob, out, S_, H_, NH_ * D_);
}
